// WindowAttention_2559800508439
// MI455X (gfx1250) — matrix operand order repaired and checked by reading, not compiled, not run on hardware
//
#include <hip/hip_runtime.h>
#include <hip/hip_bf16.h>

// CDNA5 (gfx1250) fused SwinV2 window attention, wave32 WMMA bf16 pipeline.
// GEMMs: v_wmma_f32_16x16x32_bf16, A staged once in LDS + cached in VGPRs,
// bf16 weights L2-resident. B operand uses an explicit 4-buffer modulo
// pipeline (reload of each buffer placed right after its consumer -> WAR
// hazard pins schedule, no copies for RA to coalesce). Dual accumulators.

typedef __attribute__((ext_vector_type(16))) __bf16 v16bf;
typedef __attribute__((ext_vector_type(8)))  float  v8f;

#define HEADS 12
#define HD 32
#define NTOK 64
#define DIM 384
#define NWIN 36

#define LDA   408   // x/attn-out LDS tile stride (bf16): 204 dw % 64 = 12 -> conflict-free
#define LDQ   40    // q/k LDS stride (bf16): 20 dw -> conflict-free
#define LDS_S 65    // S stride (f32): spreads softmax rows over banks
#define LDP   72    // P stride (bf16): 36 dw -> conflict-free
#define LDVT  72    // V^T stride (bf16)

__device__ __forceinline__ v8f wmma_bf16(v16bf a, v16bf b, v8f c) {
    return __builtin_amdgcn_wmma_f32_16x16x32_bf16(
        false, a, false, b, (short)0, c, false, false);
}

// A 16x32 bf16: lanes 0-15 K=0..15, lanes 16-31 K=16..31, M=lane&15
__device__ __forceinline__ v16bf load_a_bf16(const __bf16* base,
                                             int lda, int row0, int k0, int lane) {
    const int m = lane & 15, hi = lane >> 4;
    const __bf16* p = base + (size_t)(row0 + m) * lda + k0;
    v16bf a;
#pragma unroll
    for (int j = 0; j < 8; ++j) a[j]     = p[16 * hi + j];
#pragma unroll
    for (int j = 0; j < 8; ++j) a[8 + j] = p[16 * hi + 8 + j];
    return a;
}

// B 32x16 bf16 where B[k][n] = M[(n0+n)*ldm + k0+k]  (contiguous along k)
__device__ __forceinline__ v16bf load_bT_bf16(const __bf16* M,
                                              int ldm, int n0, int k0, int lane) {
    const int n = lane & 15, hi = lane >> 4;
    const __bf16* p = M + (size_t)(n0 + n) * ldm + k0 + 16 * hi;
    v16bf b;
#pragma unroll
    for (int j = 0; j < 16; ++j) b[j] = p[j];
    return b;
}

// 12-step K loop: explicit 4-buffer software pipeline, no register copies.
__device__ __forceinline__ v8f gemm_k384(const v16bf* af, const __bf16* W,
                                         int col0, int lane) {
    v8f c0 = {}, c1 = {};
    v16bf b0 = load_bT_bf16(W, DIM, col0,  0 * 32, lane);
    v16bf b1 = load_bT_bf16(W, DIM, col0,  1 * 32, lane);
    v16bf b2 = load_bT_bf16(W, DIM, col0,  2 * 32, lane);
    v16bf b3 = load_bT_bf16(W, DIM, col0,  3 * 32, lane);
    c0 = wmma_bf16(af[0], b0, c0);  b0 = load_bT_bf16(W, DIM, col0,  4 * 32, lane);
    c1 = wmma_bf16(af[1], b1, c1);  b1 = load_bT_bf16(W, DIM, col0,  5 * 32, lane);
    c0 = wmma_bf16(af[2], b2, c0);  b2 = load_bT_bf16(W, DIM, col0,  6 * 32, lane);
    c1 = wmma_bf16(af[3], b3, c1);  b3 = load_bT_bf16(W, DIM, col0,  7 * 32, lane);
    c0 = wmma_bf16(af[4], b0, c0);  b0 = load_bT_bf16(W, DIM, col0,  8 * 32, lane);
    c1 = wmma_bf16(af[5], b1, c1);  b1 = load_bT_bf16(W, DIM, col0,  9 * 32, lane);
    c0 = wmma_bf16(af[6], b2, c0);  b2 = load_bT_bf16(W, DIM, col0, 10 * 32, lane);
    c1 = wmma_bf16(af[7], b3, c1);  b3 = load_bT_bf16(W, DIM, col0, 11 * 32, lane);
    c0 = wmma_bf16(af[8],  b0, c0);
    c1 = wmma_bf16(af[9],  b1, c1);
    c0 = wmma_bf16(af[10], b2, c0);
    c1 = wmma_bf16(af[11], b3, c1);
    v8f c;
#pragma unroll
    for (int r = 0; r < 8; ++r) c[r] = c0[r] + c1[r];
    return c;
}

// ---- f32 -> bf16 weight conversion (once) ----------------------------------
__global__ __launch_bounds__(256) void cvt_bf16_kernel(const float* __restrict__ src,
                                                       __bf16* __restrict__ dst, int n4) {
    const int i = blockIdx.x * 256 + threadIdx.x;
    if (i < n4) {
        const float4 f = *(const float4*)(src + (size_t)i * 4);
        union { unsigned short u[4]; uint2 q; } pk;
        pk.u[0] = __builtin_bit_cast(unsigned short, (__bf16)f.x);
        pk.u[1] = __builtin_bit_cast(unsigned short, (__bf16)f.y);
        pk.u[2] = __builtin_bit_cast(unsigned short, (__bf16)f.z);
        pk.u[3] = __builtin_bit_cast(unsigned short, (__bf16)f.w);
        *(uint2*)(dst + (size_t)i * 4) = pk.q;
    }
}

// ---- RPE MLP -> bias table (225 x 12) --------------------------------------
__global__ __launch_bounds__(128) void rpe_table_kernel(
    const float* __restrict__ ct, const float* __restrict__ w1,
    const float* __restrict__ b1, const float* __restrict__ w2,
    float* __restrict__ table) {
    __shared__ float hid[512];
    const int i = blockIdx.x;  // 0..224
    const float c0 = ct[2 * i], c1 = ct[2 * i + 1];
    for (int hh = threadIdx.x; hh < 512; hh += blockDim.x) {
        float v = c0 * w1[2 * hh] + c1 * w1[2 * hh + 1] + b1[hh];
        hid[hh] = v > 0.f ? v : 0.f;
    }
    __syncthreads();
    if (threadIdx.x < HEADS) {
        const int h = threadIdx.x;
        float acc = 0.f;
        for (int j = 0; j < 512; ++j) acc += hid[j] * w2[h * 512 + j];
        table[i * HEADS + h] = acc;
    }
}

// ---- QKV GEMM: one block per 16-row x tile ---------------------------------
__global__ __launch_bounds__(128) void qkv_kernel(
    const float* __restrict__ x, const __bf16* __restrict__ wq,
    const float* __restrict__ q_bias, const float* __restrict__ v_bias,
    __bf16* __restrict__ qws, __bf16* __restrict__ kws, __bf16* __restrict__ vws) {
    __shared__ alignas(16) __bf16 xl[16 * LDA];
    const int row0 = blockIdx.x * 16;
    for (int i = threadIdx.x; i < (16 * DIM) / 4; i += 128) {
        const int r = i / 96, cc = (i % 96) * 4;
        const float4 f = *(const float4*)(x + (size_t)(row0 + r) * DIM + cc);
        union { unsigned short u[4]; uint2 q; } pk;
        pk.u[0] = __builtin_bit_cast(unsigned short, (__bf16)f.x);
        pk.u[1] = __builtin_bit_cast(unsigned short, (__bf16)f.y);
        pk.u[2] = __builtin_bit_cast(unsigned short, (__bf16)f.z);
        pk.u[3] = __builtin_bit_cast(unsigned short, (__bf16)f.w);
        *(uint2*)(xl + r * LDA + cc) = pk.q;
    }
    __syncthreads();

    const int lane = threadIdx.x & 31, wv = threadIdx.x >> 5;
    v16bf af[12];
#pragma unroll
    for (int k = 0; k < 12; ++k) af[k] = load_a_bf16(xl, LDA, 0, k * 32, lane);

    const int n = lane & 15, hi = lane >> 4;
    for (int ct = wv; ct < 72; ct += 4) {
        const int col0 = ct * 16;
        const int ctn = (ct + 4 < 72) ? ct + 4 : ct;   // clamped, branch-free
        __builtin_prefetch(wq + (size_t)(ctn * 16 + n) * DIM, 0, 3);

        v8f c = gemm_k384(af, wq, col0, lane);

        const int col = col0 + n;
        const float bias = (col < DIM) ? q_bias[col]
                         : (col >= 2 * DIM ? v_bias[col - 2 * DIM] : 0.f);
        const int mat = col / DIM, h = (col % DIM) >> 5, d = col & 31;
        __bf16* dst = (mat == 0) ? qws : (mat == 1) ? kws : vws;
#pragma unroll
        for (int r = 0; r < 8; ++r) {
            const int row = row0 + r + 8 * hi;
            dst[(((size_t)(row >> 6) * HEADS + h) * NTOK + (row & 63)) * HD + d] =
                (__bf16)(c[r] + bias);
        }
    }
}

// ---- Attention per (window, head) ------------------------------------------
__global__ __launch_bounds__(128) void attn_kernel(
    const __bf16* __restrict__ qws, const __bf16* __restrict__ kws,
    const __bf16* __restrict__ vws, const float* __restrict__ table,
    const int* __restrict__ rel_index, const float* __restrict__ mask,
    __bf16* __restrict__ ows) {
    __shared__ alignas(16) __bf16 ql[NTOK * LDQ];
    __shared__ alignas(16) __bf16 kl[NTOK * LDQ];
    __shared__ alignas(16) __bf16 vt[HD * LDVT];     // V transposed: [d][tok]
    __shared__ alignas(16) float  S[NTOK * LDS_S];
    __shared__ alignas(16) __bf16 P[NTOK * LDP];

    const int bh  = blockIdx.x;
    const int b   = bh / HEADS, h = bh % HEADS;
    const int wid = b % NWIN;
    const size_t base = (size_t)bh * NTOK * HD;

    for (int i = threadIdx.x; i < (NTOK * HD) / 4; i += 128) {
        const int tok = i >> 3, cc = (i & 7) * 4;
        *(uint2*)(ql + tok * LDQ + cc) = *(const uint2*)(qws + base + tok * HD + cc);
        *(uint2*)(kl + tok * LDQ + cc) = *(const uint2*)(kws + base + tok * HD + cc);
    }
    for (int i = threadIdx.x; i < NTOK * HD; i += 128) {
        const int tok = i >> 5, d = i & 31;
        vt[d * LDVT + tok] = vws[base + i];
    }
    __syncthreads();

    const int lane = threadIdx.x & 31, wv = threadIdx.x >> 5;
    const int n = lane & 15, hi = lane >> 4;

    // S = q @ k^T : wave wv owns 16-row tile, K = HD = 32 (1 WMMA per col tile)
    {
        v16bf a = load_a_bf16(ql, LDQ, wv * 16, 0, lane);
#pragma unroll
        for (int nt = 0; nt < 4; ++nt) {
            v16bf bm = load_bT_bf16(kl, LDQ, nt * 16, 0, lane);
            v8f c = {};
            c = wmma_bf16(a, bm, c);
#pragma unroll
            for (int r = 0; r < 8; ++r)
                S[(wv * 16 + r + 8 * hi) * LDS_S + nt * 16 + n] = c[r];
        }
    }
    __syncthreads();

    // scale + rel-pos bias + mask + softmax. All 128 threads: a thread pair
    // owns one row (32 cols each); pair-reduce max/sum with __shfl_xor.
    {
        const int i  = threadIdx.x >> 1;       // row 0..63
        const int j0 = (threadIdx.x & 1) * 32; // column half
        const float scale = 0.17677669529663688f;  // HD^-0.5
        float mx = -3.4e38f;
#pragma unroll 4
        for (int j = 0; j < 32; ++j) {
            const int jj = j0 + j;
            float v = S[i * LDS_S + jj] * scale
                    + table[rel_index[i * NTOK + jj] * HEADS + h]
                    + mask[((size_t)wid * NTOK + i) * NTOK + jj];
            S[i * LDS_S + jj] = v;
            mx = fmaxf(mx, v);
        }
        mx = fmaxf(mx, __shfl_xor(mx, 1, 32));
        float sum = 0.f;
#pragma unroll 4
        for (int j = 0; j < 32; ++j) {
            const int jj = j0 + j;
            float e = __expf(S[i * LDS_S + jj] - mx);
            S[i * LDS_S + jj] = e;
            sum += e;
        }
        sum += __shfl_xor(sum, 1, 32);
        const float inv = 1.f / sum;
#pragma unroll 4
        for (int j = 0; j < 32; ++j) {
            const int jj = j0 + j;
            P[i * LDP + jj] = (__bf16)(S[i * LDS_S + jj] * inv);
        }
    }
    __syncthreads();

    // O = P @ V : B[k][n] = v[k][n] = vt[n][k] -> contiguous LDS reads
#pragma unroll
    for (int nt = 0; nt < 2; ++nt) {
        v8f c = {};
#pragma unroll
        for (int kk = 0; kk < NTOK; kk += 32) {
            v16bf a  = load_a_bf16(P, LDP, wv * 16, kk, lane);
            v16bf bm = load_bT_bf16(vt, LDVT, nt * 16, kk, lane);
            c = wmma_bf16(a, bm, c);
        }
#pragma unroll
        for (int r = 0; r < 8; ++r) {
            const int tok = wv * 16 + r + 8 * hi;
            ows[((size_t)b * NTOK + tok) * DIM + h * HD + nt * 16 + n] = (__bf16)c[r];
        }
    }
}

// ---- Output projection: same A-staged, pipelined-B structure ---------------
__global__ __launch_bounds__(128) void proj_kernel(
    const __bf16* __restrict__ aws, const __bf16* __restrict__ wp,
    const float* __restrict__ proj_b, float* __restrict__ out) {
    __shared__ alignas(16) __bf16 xl[16 * LDA];
    const int row0 = blockIdx.x * 16;
    for (int i = threadIdx.x; i < (16 * DIM) / 4; i += 128) {
        const int r = i / 96, cc = (i % 96) * 4;
        *(uint2*)(xl + r * LDA + cc) = *(const uint2*)(aws + (size_t)(row0 + r) * DIM + cc);
    }
    __syncthreads();

    const int lane = threadIdx.x & 31, wv = threadIdx.x >> 5;
    v16bf af[12];
#pragma unroll
    for (int k = 0; k < 12; ++k) af[k] = load_a_bf16(xl, LDA, 0, k * 32, lane);

    const int n = lane & 15, hi = lane >> 4;
    for (int ct = wv; ct < 24; ct += 4) {
        const int col0 = ct * 16;
        const int ctn = (ct + 4 < 24) ? ct + 4 : ct;
        __builtin_prefetch(wp + (size_t)(ctn * 16 + n) * DIM, 0, 3);

        v8f c = gemm_k384(af, wp, col0, lane);

        const float bias = proj_b[col0 + n];
#pragma unroll
        for (int r = 0; r < 8; ++r) {
            const int row = row0 + r + 8 * hi;
            out[(size_t)row * DIM + col0 + n] = c[r] + bias;
        }
    }
}

extern "C" void kernel_launch(void* const* d_in, const int* in_sizes, int n_in,
                              void* d_out, int out_size, void* d_ws, size_t ws_size,
                              hipStream_t stream) {
    const float* x       = (const float*)d_in[0];
    const float* mask    = (const float*)d_in[1];
    const float* qkv_w   = (const float*)d_in[2];
    const float* q_bias  = (const float*)d_in[3];
    const float* v_bias  = (const float*)d_in[4];
    const float* rpe_w1  = (const float*)d_in[5];
    const float* rpe_b1  = (const float*)d_in[6];
    const float* rpe_w2  = (const float*)d_in[7];
    const float* proj_w  = (const float*)d_in[8];
    const float* proj_b  = (const float*)d_in[9];
    const float* coords  = (const float*)d_in[10];
    const int*   rel_idx = (const int*)d_in[11];
    float* out = (float*)d_out;

    const size_t B   = 2304;
    const size_t WQN = (size_t)3 * DIM * DIM;        // 442368
    const size_t WPN = (size_t)DIM * DIM;            // 147456
    const size_t PER = B * HEADS * NTOK * HD * sizeof(__bf16);  // 108 MiB

    char* ws = (char*)d_ws;
    float*  table = (float*)(ws + 0);
    __bf16* wqbf  = (__bf16*)(ws + 16384);
    __bf16* wpbf  = (__bf16*)(ws + 16384 + WQN * 2);
    const size_t qoff = 16384 + (WQN + WPN) * 2;     // 1196032, 256B aligned
    __bf16* qws = (__bf16*)(ws + qoff);
    __bf16* kws = (__bf16*)(ws + qoff + PER);
    __bf16* vws = (__bf16*)(ws + qoff + 2 * PER);
    __bf16* ows = (__bf16*)(ws + qoff + 3 * PER);

    cvt_bf16_kernel<<<(int)((WQN / 4 + 255) / 256), 256, 0, stream>>>(qkv_w, wqbf, (int)(WQN / 4));
    cvt_bf16_kernel<<<(int)((WPN / 4 + 255) / 256), 256, 0, stream>>>(proj_w, wpbf, (int)(WPN / 4));
    rpe_table_kernel<<<225, 128, 0, stream>>>(coords, rpe_w1, rpe_b1, rpe_w2, table);

    qkv_kernel<<<(int)((B * NTOK) / 16), 128, 0, stream>>>(x, wqbf, q_bias, v_bias,
                                                           qws, kws, vws);
    attn_kernel<<<(int)(B * HEADS), 128, 0, stream>>>(qws, kws, vws, table, rel_idx,
                                                      mask, ows);
    proj_kernel<<<(int)((B * NTOK) / 16), 128, 0, stream>>>(ows, wpbf, proj_b, out);
}
